// LFT_ResnetBlock_893353198107
// MI455X (gfx1250) — matrix operand order repaired and checked by reading, not compiled, not run on hardware
//
#include <hip/hip_runtime.h>
#include <hip/hip_bf16.h>
#include <stdint.h>

// ---------------------------------------------------------------------------
// CDNA5 (gfx1250, wave32) implementation of the LFT ResNet block.
// All 3x3/1x1 convs run as implicit GEMM on v_wmma_f32_16x16x32_f16.
// Activations live in NHWC f16 (entire working set fits in 192MB L2);
// weights are pre-packed to [tap][OCpad][Cin] f16 so every WMMA operand
// fragment is two contiguous b128 global loads per lane.
// ---------------------------------------------------------------------------

typedef __attribute__((ext_vector_type(16))) _Float16 v16h;
typedef __attribute__((ext_vector_type(8)))  _Float16 v8h;
typedef __attribute__((ext_vector_type(8)))  float    v8f;

#define HH   128
#define WWD  128
#define HWN  (HH*WWD)       // 16384 pixels per image
#define NB   2
#define PTOT (NB*HWN)       // 32768 flat pixels

// B fragment (32x16 f16): lane holds two 8-half chunks 16 halfs apart of one column.
__device__ __forceinline__ v16h ldB16(const _Float16* p) {
  v16h r;
  ((int4*)&r)[0] = ((const int4*)p)[0];
  ((int4*)&r)[1] = ((const int4*)p)[2];
  return r;
}
// A fragment (16x32 f16): lane holds two 8-half chunks 16 halfs apart.
__device__ __forceinline__ v16h ldA16(const _Float16* p) {
  v16h r;
  ((int4*)&r)[0] = *(const int4*)(p);
  ((int4*)&r)[1] = *(const int4*)(p + 16);
  return r;
}

// ---------------------------------------------------------------------------
// Weight pack: OIHW f32 -> [tap][OCp][IC] f16 (oc zero-padded to OCp)
// ---------------------------------------------------------------------------
__global__ void pack_w(const float* __restrict__ w, _Float16* __restrict__ dst,
                       int OC, int IC, int KK, int OCp) {
  int t = blockIdx.x * 256 + threadIdx.x;
  int total = KK * OCp * IC;
  if (t >= total) return;
  int kk = t / (OCp * IC);
  int r  = t % (OCp * IC);
  int oc = r / IC, ic = r % IC;
  float v = (oc < OC) ? w[((size_t)oc * IC + ic) * KK + kk] : 0.f;
  dst[t] = (_Float16)v;
}

// NCHW f32 -> NHWC f16 (one lane converts 8 consecutive channels of a pixel)
__global__ void to_nhwc(const float* __restrict__ in, _Float16* __restrict__ out, int C) {
  int t = blockIdx.x * 256 + threadIdx.x;
  int cg = C >> 3;
  if (t >= PTOT * cg) return;
  int px = t / cg;
  int c0 = (t % cg) << 3;
  int b = px >> 14, p = px & (HWN - 1);
  v8h o;
  #pragma unroll
  for (int k = 0; k < 8; ++k)
    o[k] = (_Float16)in[((size_t)(b * C + c0 + k)) * HWN + p];
  *(int4*)(out + (size_t)px * C + c0) = *(int4*)&o;
}

// ---------------------------------------------------------------------------
// Instance-norm statistics
// ---------------------------------------------------------------------------
__global__ void stats_nchw(const float* __restrict__ x,
                           float* __restrict__ mu, float* __restrict__ rs) {
  int bc = blockIdx.x;                      // b*256 + c
  const float* p = x + (size_t)bc * HWN;
  float s = 0.f, s2 = 0.f;
  for (int i = threadIdx.x; i < HWN; i += 256) { float v = p[i]; s += v; s2 += v * v; }
  __shared__ float sh[256], sh2[256];
  sh[threadIdx.x] = s; sh2[threadIdx.x] = s2;
  __syncthreads();
  for (int st = 128; st > 0; st >>= 1) {
    if (threadIdx.x < st) { sh[threadIdx.x] += sh[threadIdx.x + st]; sh2[threadIdx.x] += sh2[threadIdx.x + st]; }
    __syncthreads();
  }
  if (threadIdx.x == 0) {
    float m = sh[0] * (1.f / HWN);
    float var = sh2[0] * (1.f / HWN) - m * m;
    mu[bc] = m; rs[bc] = rsqrtf(var + 1e-5f);
  }
}

// NHWC f16 stats over channel range [0, Ccnt), tensor channel stride Cst.
// grid.x = NB * (Ccnt/8). 256 threads = 32 pixel-lanes x 8 channel-slots.
__global__ void stats_nhwc(const _Float16* __restrict__ in,
                           float* __restrict__ mu, float* __restrict__ rs,
                           int Cst, int Ccnt) {
  int ng = Ccnt >> 3;
  int b = blockIdx.x / ng;
  int c0 = (blockIdx.x % ng) << 3;
  int cof = threadIdx.x & 7, pl = threadIdx.x >> 3;
  const _Float16* base = in + (size_t)b * HWN * Cst + c0 + cof;
  float s = 0.f, s2 = 0.f;
  for (int p = pl; p < HWN; p += 32) { float v = (float)base[(size_t)p * Cst]; s += v; s2 += v * v; }
  __shared__ float sh[256], sh2[256];
  sh[threadIdx.x] = s; sh2[threadIdx.x] = s2;
  __syncthreads();
  for (int st = 128; st >= 8; st >>= 1) {
    if (threadIdx.x < st) { sh[threadIdx.x] += sh[threadIdx.x + st]; sh2[threadIdx.x] += sh2[threadIdx.x + st]; }
    __syncthreads();
  }
  if (threadIdx.x < 8) {
    float m = sh[threadIdx.x] * (1.f / HWN);
    float var = sh2[threadIdx.x] * (1.f / HWN) - m * m;
    mu[b * Ccnt + c0 + threadIdx.x] = m;
    rs[b * Ccnt + c0 + threadIdx.x] = rsqrtf(var + 1e-5f);
  }
}

// Normalize first 32 channels of actv (64ch NHWC f16) in place.
__global__ void norm_actv(_Float16* __restrict__ actv,
                          const float* __restrict__ mu, const float* __restrict__ rs) {
  int t = blockIdx.x * 256 + threadIdx.x;
  if (t >= PTOT * 4) return;                // 32 channels / 8 per thread
  int px = t >> 2;
  int c0 = (t & 3) << 3;
  int b = px >> 14;
  v8h v = *(v8h*)(actv + (size_t)px * 64 + c0);
  #pragma unroll
  for (int k = 0; k < 8; ++k) {
    int c = c0 + k;
    float f = ((float)v[k] - mu[b * 32 + c]) * rs[b * 32 + c];
    v[k] = (_Float16)f;
  }
  *(int4*)(actv + (size_t)px * 64 + c0) = *(int4*)&v;
}

// ---------------------------------------------------------------------------
// Group attention: actv[px, g*8+gc] = sum_j wgt[px, g*9+j] * a1[nbr_j(px), g*8+gc]
// ---------------------------------------------------------------------------
__global__ void attn_kernel(const _Float16* __restrict__ a1,   // [P,64]
                            const _Float16* __restrict__ wgt,  // [P,80] (72 live)
                            _Float16* __restrict__ actv) {     // [P,64]
  int t = blockIdx.x * 256 + threadIdx.x;
  if (t >= PTOT * 8) return;
  int px = t >> 3, g = t & 7;
  int p = px & (HWN - 1);
  int h = p >> 7, w = p & 127;
  int nbase = px - p;                       // b*HWN
  float acc[8] = {0.f,0.f,0.f,0.f,0.f,0.f,0.f,0.f};
  #pragma unroll
  for (int ky = 0; ky < 3; ++ky) {
    int y = h + ky - 1;
    #pragma unroll
    for (int kx = 0; kx < 3; ++kx) {
      int xx = w + kx - 1;
      if ((unsigned)y < (unsigned)HH && (unsigned)xx < (unsigned)WWD) {
        int j = ky * 3 + kx;
        float wv = (float)wgt[(size_t)px * 80 + g * 9 + j];
        v8h av = *(const v8h*)(a1 + ((size_t)(nbase + y * WWD + xx)) * 64 + g * 8);
        #pragma unroll
        for (int k = 0; k < 8; ++k) acc[k] += wv * (float)av[k];
      }
    }
  }
  v8h o;
  #pragma unroll
  for (int k = 0; k < 8; ++k) o[k] = (_Float16)acc[k];
  *(int4*)(actv + (size_t)px * 64 + g * 8) = *(int4*)&o;
}

// ---------------------------------------------------------------------------
// Generic conv (implicit GEMM, WMMA f16->f32). Wave tile: 16 oc x 64 px.
// in : NHWC f16 [P, Cin], apk : [tap][OCp][Cin] f16, out : NHWC f16 [P, OCst]
// ---------------------------------------------------------------------------
__global__ void __launch_bounds__(256)
conv_wmma(const _Float16* __restrict__ in, const _Float16* __restrict__ apk,
          const float* __restrict__ bias, _Float16* __restrict__ out,
          int Cin, int OCp, int OCreal, int OCst, int ksz, int relu, int total_waves) {
  int wid = threadIdx.x >> 5, lane = threadIdx.x & 31;
  int gw = blockIdx.x * 8 + wid;
  if (gw >= total_waves) return;            // wave-uniform exit
  int OT = OCp >> 4;
  int ot = gw % OT, pt = gw / OT;
  int p0 = pt << 6;
  int b  = p0 >> 14;
  int hh = (p0 >> 7) & 127;
  int w0 = p0 & 127;
  int oc0 = ot << 4;
  int lo = lane & 15, hi = lane >> 4;
  v8f ac0 = {}, ac1 = {}, ac2 = {}, ac3 = {};
  int off = (ksz == 3) ? 1 : 0;
  int kk2 = ksz * ksz;
  for (int kk = 0; kk < kk2; ++kk) {
    int y  = hh + (kk / ksz) - off;
    int x0 = w0 + lo + (kk % ksz) - off;
    bool rowok = ((unsigned)y < (unsigned)HH);
    const _Float16* aP = apk + ((size_t)kk * OCp + oc0 + lo) * Cin + hi * 8;
    if (kk + 1 < kk2)                       // warm next weight tap (global_prefetch_b8)
      __builtin_prefetch(apk + ((size_t)(kk + 1) * OCp + oc0 + lo) * Cin, 0, 3);
    const _Float16* rB = in + ((size_t)b * HWN + (rowok ? y : 0) * WWD) * Cin + hi * 8;
    for (int c0 = 0; c0 < Cin; c0 += 32) {
      v16h af = ldA16(aP + c0);
      v16h b0 = {}, b1 = {}, b2 = {}, b3 = {};
      if (rowok) {
        int x1 = x0 + 16, x2 = x0 + 32, x3 = x0 + 48;
        if ((unsigned)x0 < (unsigned)WWD) b0 = ldB16(rB + (size_t)x0 * Cin + c0);
        if ((unsigned)x1 < (unsigned)WWD) b1 = ldB16(rB + (size_t)x1 * Cin + c0);
        if ((unsigned)x2 < (unsigned)WWD) b2 = ldB16(rB + (size_t)x2 * Cin + c0);
        if ((unsigned)x3 < (unsigned)WWD) b3 = ldB16(rB + (size_t)x3 * Cin + c0);
      }
      ac0 = __builtin_amdgcn_wmma_f32_16x16x32_f16(false, af, false, b0, (short)0, ac0, false, false);
      ac1 = __builtin_amdgcn_wmma_f32_16x16x32_f16(false, af, false, b1, (short)0, ac1, false, false);
      ac2 = __builtin_amdgcn_wmma_f32_16x16x32_f16(false, af, false, b2, (short)0, ac2, false, false);
      ac3 = __builtin_amdgcn_wmma_f32_16x16x32_f16(false, af, false, b3, (short)0, ac3, false, false);
    }
  }
  int ocb = oc0 + hi * 8;
  float bs[8];
  #pragma unroll
  for (int r = 0; r < 8; ++r) bs[r] = (ocb + r < OCreal) ? bias[ocb + r] : 0.f;
#define STORE_FRAG(F, ACC) { \
    size_t px = (size_t)p0 + (F) * 16 + lo; \
    v8h hv; \
    for (int r = 0; r < 8; ++r) { \
      float v = ACC[r] + bs[r]; \
      if (relu) v = v > 0.f ? v : 0.f; \
      hv[r] = (_Float16)v; \
    } \
    *(int4*)(out + px * OCst + ocb) = *(int4*)&hv; }
  STORE_FRAG(0, ac0) STORE_FRAG(1, ac1) STORE_FRAG(2, ac2) STORE_FRAG(3, ac3)
#undef STORE_FRAG
}

// ---------------------------------------------------------------------------
// Fused gamma/beta conv (64->256 x2) + IN-apply + leaky-ReLU epilogue.
// srcN: x NCHW f32 (phase 0)  OR  srcH: dx NHWC f16 (phase 1)
// ---------------------------------------------------------------------------
__global__ void __launch_bounds__(256)
conv_gb(const _Float16* __restrict__ in,   // actv2 [P,64]
        const _Float16* __restrict__ apkG, const _Float16* __restrict__ apkB,
        const float* __restrict__ biasG, const float* __restrict__ biasB,
        const float* __restrict__ srcN, const _Float16* __restrict__ srcH,
        const float* __restrict__ mu, const float* __restrict__ rs,
        _Float16* __restrict__ out, int total_waves) {
  const int Cin = 64, OCp = 256;
  int wid = threadIdx.x >> 5, lane = threadIdx.x & 31;
  int gw = blockIdx.x * 8 + wid;
  if (gw >= total_waves) return;
  int ot = gw % 16, pt = gw / 16;
  int p0 = pt << 6;
  int b  = p0 >> 14;
  int hh = (p0 >> 7) & 127;
  int w0 = p0 & 127;
  int oc0 = ot << 4;
  int lo = lane & 15, hi = lane >> 4;
  v8f g0 = {}, g1 = {}, g2 = {}, g3 = {};
  v8f q0 = {}, q1 = {}, q2 = {}, q3 = {};
  for (int kk = 0; kk < 9; ++kk) {
    int y  = hh + (kk / 3) - 1;
    int x0 = w0 + lo + (kk % 3) - 1;
    bool rowok = ((unsigned)y < (unsigned)HH);
    const _Float16* aPG = apkG + ((size_t)kk * OCp + oc0 + lo) * Cin + hi * 8;
    const _Float16* aPB = apkB + ((size_t)kk * OCp + oc0 + lo) * Cin + hi * 8;
    const _Float16* rB = in + ((size_t)b * HWN + (rowok ? y : 0) * WWD) * Cin + hi * 8;
    for (int c0 = 0; c0 < Cin; c0 += 32) {
      v16h afG = ldA16(aPG + c0);
      v16h afB = ldA16(aPB + c0);
      v16h b0 = {}, b1 = {}, b2 = {}, b3 = {};
      if (rowok) {
        int x1 = x0 + 16, x2 = x0 + 32, x3 = x0 + 48;
        if ((unsigned)x0 < (unsigned)WWD) b0 = ldB16(rB + (size_t)x0 * Cin + c0);
        if ((unsigned)x1 < (unsigned)WWD) b1 = ldB16(rB + (size_t)x1 * Cin + c0);
        if ((unsigned)x2 < (unsigned)WWD) b2 = ldB16(rB + (size_t)x2 * Cin + c0);
        if ((unsigned)x3 < (unsigned)WWD) b3 = ldB16(rB + (size_t)x3 * Cin + c0);
      }
      g0 = __builtin_amdgcn_wmma_f32_16x16x32_f16(false, afG, false, b0, (short)0, g0, false, false);
      g1 = __builtin_amdgcn_wmma_f32_16x16x32_f16(false, afG, false, b1, (short)0, g1, false, false);
      g2 = __builtin_amdgcn_wmma_f32_16x16x32_f16(false, afG, false, b2, (short)0, g2, false, false);
      g3 = __builtin_amdgcn_wmma_f32_16x16x32_f16(false, afG, false, b3, (short)0, g3, false, false);
      q0 = __builtin_amdgcn_wmma_f32_16x16x32_f16(false, afB, false, b0, (short)0, q0, false, false);
      q1 = __builtin_amdgcn_wmma_f32_16x16x32_f16(false, afB, false, b1, (short)0, q1, false, false);
      q2 = __builtin_amdgcn_wmma_f32_16x16x32_f16(false, afB, false, b2, (short)0, q2, false, false);
      q3 = __builtin_amdgcn_wmma_f32_16x16x32_f16(false, afB, false, b3, (short)0, q3, false, false);
    }
  }
  int ocb = oc0 + hi * 8;
  float bg[8], bb[8], sm[8], sr[8];
  #pragma unroll
  for (int r = 0; r < 8; ++r) {
    bg[r] = biasG[ocb + r]; bb[r] = biasB[ocb + r];
    sm[r] = mu[b * 256 + ocb + r]; sr[r] = rs[b * 256 + ocb + r];
  }
#define GB_EPI(F, AG, AB) { \
    size_t px = (size_t)p0 + (F) * 16 + lo; \
    int p = (int)(px & (HWN - 1)); \
    float xv[8]; \
    if (srcN) { \
      for (int r = 0; r < 8; ++r) xv[r] = srcN[((size_t)(b * 256 + ocb + r)) * HWN + p]; \
    } else { \
      v8h xh = *(const v8h*)(srcH + px * 256 + ocb); \
      for (int r = 0; r < 8; ++r) xv[r] = (float)xh[r]; \
    } \
    v8h hv; \
    for (int r = 0; r < 8; ++r) { \
      float xn = (xv[r] - sm[r]) * sr[r]; \
      float v = xn * (1.f + AG[r] + bg[r]) + (AB[r] + bb[r]); \
      v = v > 0.f ? v : 0.2f * v; \
      hv[r] = (_Float16)v; \
    } \
    *(int4*)(out + px * 256 + ocb) = *(int4*)&hv; }
  GB_EPI(0, g0, q0) GB_EPI(1, g1, q1) GB_EPI(2, g2, q2) GB_EPI(3, g3, q3)
#undef GB_EPI
}

// ---------------------------------------------------------------------------
// Fused gated conv (256->256 x2) + c*sigmoid(m) epilogue.
// outH != null : store NHWC f16 (dx1).  Else: outF = resid + val, NCHW f32.
// ---------------------------------------------------------------------------
__global__ void __launch_bounds__(256)
conv_gated(const _Float16* __restrict__ in,  // h [P,256]
           const _Float16* __restrict__ apkC, const _Float16* __restrict__ apkM,
           const float* __restrict__ biasC, const float* __restrict__ biasM,
           const float* __restrict__ resid, _Float16* __restrict__ outH,
           float* __restrict__ outF, int total_waves) {
  const int Cin = 256, OCp = 256;
  int wid = threadIdx.x >> 5, lane = threadIdx.x & 31;
  int gw = blockIdx.x * 8 + wid;
  if (gw >= total_waves) return;
  int ot = gw % 16, pt = gw / 16;
  int p0 = pt << 6;
  int b  = p0 >> 14;
  int hh = (p0 >> 7) & 127;
  int w0 = p0 & 127;
  int oc0 = ot << 4;
  int lo = lane & 15, hi = lane >> 4;
  v8f c0a = {}, c1a = {}, c2a = {}, c3a = {};
  v8f m0a = {}, m1a = {}, m2a = {}, m3a = {};
  for (int kk = 0; kk < 9; ++kk) {
    int y  = hh + (kk / 3) - 1;
    int x0 = w0 + lo + (kk % 3) - 1;
    bool rowok = ((unsigned)y < (unsigned)HH);
    const _Float16* aPC = apkC + ((size_t)kk * OCp + oc0 + lo) * Cin + hi * 8;
    const _Float16* aPM = apkM + ((size_t)kk * OCp + oc0 + lo) * Cin + hi * 8;
    const _Float16* rB = in + ((size_t)b * HWN + (rowok ? y : 0) * WWD) * Cin + hi * 8;
    for (int cc = 0; cc < Cin; cc += 32) {
      v16h afC = ldA16(aPC + cc);
      v16h afM = ldA16(aPM + cc);
      v16h b0 = {}, b1 = {}, b2 = {}, b3 = {};
      if (rowok) {
        int x1 = x0 + 16, x2 = x0 + 32, x3 = x0 + 48;
        if ((unsigned)x0 < (unsigned)WWD) b0 = ldB16(rB + (size_t)x0 * Cin + cc);
        if ((unsigned)x1 < (unsigned)WWD) b1 = ldB16(rB + (size_t)x1 * Cin + cc);
        if ((unsigned)x2 < (unsigned)WWD) b2 = ldB16(rB + (size_t)x2 * Cin + cc);
        if ((unsigned)x3 < (unsigned)WWD) b3 = ldB16(rB + (size_t)x3 * Cin + cc);
      }
      c0a = __builtin_amdgcn_wmma_f32_16x16x32_f16(false, afC, false, b0, (short)0, c0a, false, false);
      c1a = __builtin_amdgcn_wmma_f32_16x16x32_f16(false, afC, false, b1, (short)0, c1a, false, false);
      c2a = __builtin_amdgcn_wmma_f32_16x16x32_f16(false, afC, false, b2, (short)0, c2a, false, false);
      c3a = __builtin_amdgcn_wmma_f32_16x16x32_f16(false, afC, false, b3, (short)0, c3a, false, false);
      m0a = __builtin_amdgcn_wmma_f32_16x16x32_f16(false, afM, false, b0, (short)0, m0a, false, false);
      m1a = __builtin_amdgcn_wmma_f32_16x16x32_f16(false, afM, false, b1, (short)0, m1a, false, false);
      m2a = __builtin_amdgcn_wmma_f32_16x16x32_f16(false, afM, false, b2, (short)0, m2a, false, false);
      m3a = __builtin_amdgcn_wmma_f32_16x16x32_f16(false, afM, false, b3, (short)0, m3a, false, false);
    }
  }
  int ocb = oc0 + hi * 8;
  float bc[8], bm[8];
  #pragma unroll
  for (int r = 0; r < 8; ++r) { bc[r] = biasC[ocb + r]; bm[r] = biasM[ocb + r]; }
#define GATED_EPI(F, AC, AM) { \
    size_t px = (size_t)p0 + (F) * 16 + lo; \
    int p = (int)(px & (HWN - 1)); \
    float vals[8]; \
    for (int r = 0; r < 8; ++r) { \
      float cv = AC[r] + bc[r]; \
      float mv = AM[r] + bm[r]; \
      vals[r] = cv * (1.f / (1.f + __expf(-mv))); \
    } \
    if (outH) { \
      v8h hv; \
      for (int r = 0; r < 8; ++r) hv[r] = (_Float16)vals[r]; \
      *(int4*)(outH + px * 256 + ocb) = *(int4*)&hv; \
    } else { \
      for (int r = 0; r < 8; ++r) { \
        size_t o = ((size_t)(b * 256 + ocb + r)) * HWN + p; \
        outF[o] = resid[o] + vals[r]; \
      } \
    } }
  GATED_EPI(0, c0a, m0a) GATED_EPI(1, c1a, m1a) GATED_EPI(2, c2a, m2a) GATED_EPI(3, c3a, m3a)
#undef GATED_EPI
}

// ---------------------------------------------------------------------------
// Host orchestration
// ---------------------------------------------------------------------------
extern "C" void kernel_launch(void* const* d_in, const int* in_sizes, int n_in,
                              void* d_out, int out_size, void* d_ws, size_t ws_size,
                              hipStream_t stream) {
  (void)in_sizes; (void)n_in; (void)out_size;
  const float* x   = (const float*)d_in[0];
  const float* seg = (const float*)d_in[1];
  // setup_inputs() dict order: x, seg, params{norm0, conv0, norm1, conv1}
  // lft block: s0w,s0b,s1w,s1b,c1w,c1b,c2w,c2b,gw,gb,bw,bb ; gated: cw,cb,mw,mb
  const int N0 = 2, C0 = 14, N1 = 18, C1 = 30;

  char* ws = (char*)d_ws;
  size_t off = 0;
  auto alloc = [&](size_t bytes) -> char* {
    char* p = ws + off;
    off += (bytes + 255) & ~(size_t)255;
    return p;
  };
  _Float16* segH = (_Float16*)alloc((size_t)PTOT * 256 * 2);
  _Float16* xH   = (_Float16*)alloc((size_t)PTOT * 256 * 2);  // reused as dxH after phase-0 s0
  _Float16* dxH  = xH;
  _Float16* hH   = (_Float16*)alloc((size_t)PTOT * 256 * 2);
  _Float16* a0B  = (_Float16*)alloc((size_t)PTOT * 64 * 2);
  _Float16* a1B  = (_Float16*)alloc((size_t)PTOT * 64 * 2);
  _Float16* t1B  = (_Float16*)alloc((size_t)PTOT * 32 * 2);
  _Float16* wgB  = (_Float16*)alloc((size_t)PTOT * 80 * 2);
  _Float16* acB  = (_Float16*)alloc((size_t)PTOT * 64 * 2);
  float* muX = (float*)alloc(512 * 4); float* rsX = (float*)alloc(512 * 4);
  float* muA = (float*)alloc(64 * 4);  float* rsA = (float*)alloc(64 * 4);
  float* muD = (float*)alloc(512 * 4); float* rsD = (float*)alloc(512 * 4);
  _Float16 *pk_s0[2], *pk_s1[2], *pk_c1[2], *pk_c2[2], *pk_g[2], *pk_b[2], *pk_c[2], *pk_m[2];
  for (int p = 0; p < 2; ++p) {
    pk_s0[p] = (_Float16*)alloc((size_t)9 * 64 * 256 * 2);
    pk_s1[p] = (_Float16*)alloc((size_t)9 * 64 * 256 * 2);
    pk_c1[p] = (_Float16*)alloc((size_t)1 * 32 * 64 * 2);
    pk_c2[p] = (_Float16*)alloc((size_t)1 * 80 * 32 * 2);
    pk_g[p]  = (_Float16*)alloc((size_t)9 * 256 * 64 * 2);
    pk_b[p]  = (_Float16*)alloc((size_t)9 * 256 * 64 * 2);
    pk_c[p]  = (_Float16*)alloc((size_t)9 * 256 * 256 * 2);
    pk_m[p]  = (_Float16*)alloc((size_t)9 * 256 * 256 * 2);
  }
  if (off > ws_size) return;   // insufficient workspace

  auto packW = [&](const void* w, _Float16* dst, int OC, int IC, int KK, int OCp) {
    int total = KK * OCp * IC;
    pack_w<<<(total + 255) / 256, 256, 0, stream>>>((const float*)w, dst, OC, IC, KK, OCp);
  };
  for (int p = 0; p < 2; ++p) {
    int nb = p ? N1 : N0, cb = p ? C1 : C0;
    packW(d_in[nb + 0],  pk_s0[p], 64, 256, 9, 64);
    packW(d_in[nb + 2],  pk_s1[p], 64, 256, 9, 64);
    packW(d_in[nb + 4],  pk_c1[p], 32, 64, 1, 32);
    packW(d_in[nb + 6],  pk_c2[p], 72, 32, 1, 80);
    packW(d_in[nb + 8],  pk_g[p], 256, 64, 9, 256);
    packW(d_in[nb + 10], pk_b[p], 256, 64, 9, 256);
    packW(d_in[cb + 0],  pk_c[p], 256, 256, 9, 256);
    packW(d_in[cb + 2],  pk_m[p], 256, 256, 9, 256);
  }
  {
    int total = PTOT * (256 / 8);
    to_nhwc<<<(total + 255) / 256, 256, 0, stream>>>(x, xH, 256);
    to_nhwc<<<(total + 255) / 256, 256, 0, stream>>>(seg, segH, 256);
  }
  stats_nchw<<<NB * 256, 256, 0, stream>>>(x, muX, rsX);

  auto conv = [&](const _Float16* in, const _Float16* apk, const void* bias,
                  _Float16* out, int Cin, int OCp, int OCreal, int OCst, int ksz, int relu) {
    int total = (PTOT / 64) * (OCp / 16);
    conv_wmma<<<(total + 7) / 8, 256, 0, stream>>>(
        in, apk, (const float*)bias, out, Cin, OCp, OCreal, OCst, ksz, relu, total);
  };

  for (int p = 0; p < 2; ++p) {
    int nb = p ? N1 : N0, cb = p ? C1 : C0;
    const _Float16* src = p ? (const _Float16*)dxH : (const _Float16*)xH;
    conv(src,  pk_s0[p], d_in[nb + 1], a0B, 256, 64, 64, 64, 3, 1);   // a0 = relu(s0(x))
    conv(segH, pk_s1[p], d_in[nb + 3], a1B, 256, 64, 64, 64, 3, 1);   // a1 = relu(s1(seg))
    conv(a0B,  pk_c1[p], d_in[nb + 5], t1B, 64, 32, 32, 32, 1, 0);    // 1x1 64->32
    conv(t1B,  pk_c2[p], d_in[nb + 7], wgB, 32, 80, 72, 80, 1, 0);    // 1x1 32->72 (pad 80)
    attn_kernel<<<PTOT * 8 / 256, 256, 0, stream>>>(a1B, wgB, acB);
    stats_nhwc<<<NB * (32 / 8), 256, 0, stream>>>(acB, muA, rsA, 64, 32);
    norm_actv<<<PTOT * 4 / 256, 256, 0, stream>>>(acB, muA, rsA);
    if (p) stats_nhwc<<<NB * (256 / 8), 256, 0, stream>>>(dxH, muD, rsD, 256, 256);
    int total = (PTOT / 64) * 16;
    conv_gb<<<(total + 7) / 8, 256, 0, stream>>>(
        acB, pk_g[p], pk_b[p], (const float*)d_in[nb + 9], (const float*)d_in[nb + 11],
        p ? (const float*)nullptr : x, p ? (const _Float16*)dxH : (const _Float16*)nullptr,
        p ? muD : muX, p ? rsD : rsX, hH, total);
    conv_gated<<<(total + 7) / 8, 256, 0, stream>>>(
        hH, pk_c[p], pk_m[p], (const float*)d_in[cb + 1], (const float*)d_in[cb + 3],
        p ? x : (const float*)nullptr,
        p ? (_Float16*)nullptr : dxH,
        p ? (float*)d_out : (float*)nullptr, total);
  }
}
